// SMoE_23983097381214
// MI455X (gfx1250) — matrix operand order repaired and checked by reading, not compiled, not run on hardware
//
#include <hip/hip_runtime.h>

typedef __attribute__((ext_vector_type(16))) __bf16 v16bf;
typedef __attribute__((ext_vector_type(8)))  float  v8f;

#define B_   32
#define S_   512
#define D_   768
#define O_   768
#define E_   8

#define BM   128
#define BN   128
#define BK   64
#define LDST 33   // dword stride for a 64-bf16 row (32 + 1 pad, avoids bank conflicts)

__device__ __forceinline__ unsigned pack_bf16x2(float lo, float hi) {
    unsigned a = __float_as_uint(lo);
    unsigned b = __float_as_uint(hi);
    a += 0x7FFFu + ((a >> 16) & 1u);   // round-to-nearest-even to bf16
    b += 0x7FFFu + ((b >> 16) & 1u);
    return (a >> 16) | (b & 0xFFFF0000u);
}

// ---------------------------------------------------------------- pooling ---
__global__ void pool_kernel(const float* __restrict__ x, float* __restrict__ pooled) {
    const int b = blockIdx.x;
    const int d = blockIdx.y * 256 + threadIdx.x;
    const float* p = x + (size_t)b * S_ * D_ + d;
    float s = 0.f;
    #pragma unroll 8
    for (int t = 0; t < S_; ++t) s += p[(size_t)t * D_];
    pooled[b * D_ + d] = s * (1.0f / (float)S_);
}

// --------------------------------------------------------- gating + loss ---
__global__ void gate_kernel(const float* __restrict__ pooled,
                            const float* __restrict__ w_gate,
                            int*   __restrict__ eidx,
                            float* __restrict__ gval,
                            float* __restrict__ loss_out) {
    __shared__ float logits[B_][E_];
    __shared__ float gates[B_][E_];
    __shared__ float imp[E_], ldv[E_];
    const int t = threadIdx.x;          // 256 threads = 32 batches x 8 experts
    {
        const int b = t >> 3, e = t & 7;
        float acc = 0.f;
        for (int d = 0; d < D_; ++d) acc += pooled[b * D_ + d] * w_gate[d * E_ + e];
        logits[b][e] = acc;
    }
    __syncthreads();
    if (t < B_) {
        float v0 = -INFINITY, v1 = -INFINITY; int i0 = 0, i1 = 0;
        for (int j = 0; j < E_; ++j) {
            float v = logits[t][j];
            if (v > v0)      { v1 = v0; i1 = i0; v0 = v; i0 = j; }
            else if (v > v1) { v1 = v; i1 = j; }
        }
        // softmax over the top-2 (v0 is the max)
        float x0 = expf(0.0f), x1 = expf(v1 - v0);
        float inv = 1.0f / (x0 + x1);
        float g0 = x0 * inv, g1 = x1 * inv;
        for (int j = 0; j < E_; ++j) gates[t][j] = 0.f;
        gates[t][i0] = g0; gates[t][i1] = g1;
        eidx[t * 2 + 0] = i0; eidx[t * 2 + 1] = i1;
        gval[t * 2 + 0] = g0; gval[t * 2 + 1] = g1;
    }
    __syncthreads();
    if (t < E_) {
        float si = 0.f, sl = 0.f;
        for (int bb = 0; bb < B_; ++bb) {
            float g = gates[bb][t];
            si += g;
            sl += (g > 0.f) ? 1.f : 0.f;
        }
        imp[t] = si; ldv[t] = sl;
    }
    __syncthreads();
    if (t == 0) {
        float mi = 0.f, ml = 0.f;
        for (int j = 0; j < E_; ++j) { mi += imp[j]; ml += ldv[j]; }
        mi *= (1.0f / E_); ml *= (1.0f / E_);
        float vi = 0.f, vl = 0.f;
        for (int j = 0; j < E_; ++j) {
            float a = imp[j] - mi; vi += a * a;
            float c = ldv[j] - ml; vl += c * c;
        }
        vi *= (1.0f / (E_ - 1)); vl *= (1.0f / (E_ - 1));
        float cv = vi / (mi * mi + 1e-10f) + vl / (ml * ml + 1e-10f);
        loss_out[0] = cv * 0.01f;
    }
}

// --------------------------------------------- fused expert-mix bf16 GEMM ---
__global__ void __launch_bounds__(256)
moe_gemm_kernel(const float* __restrict__ x,
                const float* __restrict__ weight,
                const float* __restrict__ bias,
                const int*   __restrict__ eidx,
                const float* __restrict__ gval,
                float* __restrict__ y) {
    __shared__ unsigned lA[BM * LDST];   // X tile   (BM rows x 64 bf16)
    __shared__ unsigned lB[BN * LDST];   // Wmix^T   (BN o-rows x 64 bf16 along d)
    __shared__ float    eb[BN];          // mixed bias slice

    const int tid  = threadIdx.x;
    const int lane = tid & 31;
    const int wave = tid >> 5;     // 0..7
    const int wm   = wave >> 1;    // 0..3  M-direction wave
    const int wn   = wave & 1;     // 0..1  N-direction wave
    const int half = lane >> 4;    // hi/lo lane group
    const int l16  = lane & 15;

    const int b     = blockIdx.z;
    const int oBase = blockIdx.x * BN;
    const int sBase = blockIdx.y * BM;

    const int   e0 = eidx[b * 2 + 0];
    const int   e1 = eidx[b * 2 + 1];
    const float g0 = gval[b * 2 + 0];
    const float g1 = gval[b * 2 + 1];

    const float* w0 = weight + (size_t)e0 * O_ * D_;
    const float* w1 = weight + (size_t)e1 * O_ * D_;
    const float* xb = x + (size_t)b * S_ * D_;

    if (tid < BN) {
        int o = oBase + tid;
        eb[tid] = g0 * bias[e0 * O_ + o] + g1 * bias[e1 * O_ + o];
    }

    union Frag { v16bf v; unsigned u[8]; };
    const v8f vzero = {};
    v8f acc[2][4];
    #pragma unroll
    for (int i = 0; i < 2; ++i)
        #pragma unroll
        for (int j = 0; j < 4; ++j) acc[i][j] = vzero;

    for (int kk = 0; kk < D_; kk += BK) {
        // ---- stage A tile: BM x 64 fp32 -> packed bf16 LDS ----
        #pragma unroll
        for (int i = 0; i < 8; ++i) {
            int slot = tid + i * 256;
            int row  = slot >> 4;
            int c4   = slot & 15;
            const float4 v = *(const float4*)(xb + (size_t)(sBase + row) * D_ + kk + c4 * 4);
            lA[row * LDST + c4 * 2 + 0] = pack_bf16x2(v.x, v.y);
            lA[row * LDST + c4 * 2 + 1] = pack_bf16x2(v.z, v.w);
        }
        // ---- stage B^T tile: mix two experts on the fly ----
        #pragma unroll
        for (int i = 0; i < 8; ++i) {
            int slot = tid + i * 256;
            int row  = slot >> 4;            // o within tile
            int c4   = slot & 15;
            size_t off = (size_t)(oBase + row) * D_ + kk + c4 * 4;
            const float4 a = *(const float4*)(w0 + off);
            const float4 c = *(const float4*)(w1 + off);
            lB[row * LDST + c4 * 2 + 0] = pack_bf16x2(g0 * a.x + g1 * c.x, g0 * a.y + g1 * c.y);
            lB[row * LDST + c4 * 2 + 1] = pack_bf16x2(g0 * a.z + g1 * c.z, g0 * a.w + g1 * c.w);
        }
        // prefetch next K-chunk (lowers to global_prefetch_b8)
        if (kk + BK < D_) {
            int row = tid >> 1, c = (tid & 1) * 32;
            __builtin_prefetch(xb + (size_t)(sBase + row) * D_ + kk + BK + c, 0, 1);
            __builtin_prefetch(w0 + (size_t)(oBase + row) * D_ + kk + BK + c, 0, 1);
            __builtin_prefetch(w1 + (size_t)(oBase + row) * D_ + kk + BK + c, 0, 1);
        }
        __syncthreads();

        #pragma unroll
        for (int ks = 0; ks < BK; ks += 32) {
            Frag a[2], bf[4];
            // A fragments: 16x32, lanes 0-15 hold K {0-7,16-23}, lanes 16-31 hold K {8-15,24-31}
            #pragma unroll
            for (int mt = 0; mt < 2; ++mt) {
                const unsigned* base = &lA[(wm * 32 + mt * 16 + l16) * LDST];
                #pragma unroll
                for (int j = 0; j < 4; ++j) {
                    a[mt].u[j]     = base[(ks + half * 8 + 2 * j) >> 1];
                    a[mt].u[4 + j] = base[(ks + 16 + half * 8 + 2 * j) >> 1];
                }
            }
            // B fragments: 32x16, N = lane&15, lanes 0-15 K {0-7,16-23}, lanes 16-31 K {8-15,24-31} (pairs lo/hi)
            #pragma unroll
            for (int nt = 0; nt < 4; ++nt) {
                const unsigned* base = &lB[(wn * 64 + nt * 16 + l16) * LDST];
                #pragma unroll
                for (int j = 0; j < 8; ++j)
                    bf[nt].u[j] = base[(ks + (j < 4 ? half * 8 + 2 * j : 16 + half * 8 + 2 * (j - 4))) >> 1];
            }
            #pragma unroll
            for (int mt = 0; mt < 2; ++mt)
                #pragma unroll
                for (int nt = 0; nt < 4; ++nt)
                    acc[mt][nt] = __builtin_amdgcn_wmma_f32_16x16x32_bf16(
                        false, a[mt].v, false, bf[nt].v,
                        (short)0, acc[mt][nt], false, false);
        }
        __syncthreads();
    }

    // ---- store: C/D layout — VGPR r holds row M = r + 8*half, N = lane&15 ----
    #pragma unroll
    for (int mt = 0; mt < 2; ++mt) {
        #pragma unroll
        for (int nt = 0; nt < 4; ++nt) {
            int oLoc = wn * 64 + nt * 16 + l16;
            int o    = oBase + oLoc;
            float bo = eb[oLoc];
            #pragma unroll
            for (int r = 0; r < 8; ++r) {
                int s = sBase + wm * 32 + mt * 16 + r + 8 * half;
                y[(size_t)b * S_ * O_ + (size_t)s * O_ + o] = acc[mt][nt][r] + bo;
            }
        }
    }
}

// ----------------------------------------------------------------- launch ---
extern "C" void kernel_launch(void* const* d_in, const int* in_sizes, int n_in,
                              void* d_out, int out_size, void* d_ws, size_t ws_size,
                              hipStream_t stream) {
    const float* x      = (const float*)d_in[0];
    const float* w_gate = (const float*)d_in[1];
    const float* weight = (const float*)d_in[2];
    const float* bias   = (const float*)d_in[3];

    float* y    = (float*)d_out;
    float* loss = y + (size_t)B_ * S_ * O_;     // scalar after the y tensor

    float* pooled = (float*)d_ws;               // 32*768 floats
    int*   eidx   = (int*)(pooled + B_ * D_);   // 32*2 ints
    float* gval   = (float*)(eidx + B_ * 2);    // 32*2 floats

    pool_kernel<<<dim3(B_, D_ / 256), 256, 0, stream>>>(x, pooled);
    gate_kernel<<<1, 256, 0, stream>>>(pooled, w_gate, eidx, gval, loss);
    moe_gemm_kernel<<<dim3(O_ / BN, S_ / BM, B_), 256, 0, stream>>>(
        x, weight, bias, eidx, gval, y);
}
